// DCNv4_45973329936714
// MI455X (gfx1250) — matrix operand order repaired and checked by reading, not compiled, not run on hardware
//
#include <hip/hip_runtime.h>

// ---------------- problem constants (match reference) ----------------
#define CCH   64      // C
#define GRP   4       // G
#define GCH   16      // GC = C/G
#define KPTS  9       // KS*KS
#define OMD   112     // OM_DIM (ceil(4*9*3/8)*8)
// N=4, H=W=128 passed at launch

// ---------------- WMMA vector types ----------------
typedef __attribute__((ext_vector_type(16))) __bf16 bf16x16;
typedef __attribute__((ext_vector_type(8)))  float  floatx8;

union V16BF {
    bf16x16        v;
    uint4          q[2];
    unsigned short s[16];
};

// fp32 -> bf16 bits, round-to-nearest-even
__device__ __forceinline__ unsigned short f2bf_bits(float f) {
    unsigned u = __float_as_uint(f);
    unsigned r = u + 0x7FFFu + ((u >> 16) & 1u);
    return (unsigned short)(r >> 16);
}

// A-matrix (16x32 bf16) fragment load per ISA 7.12.2:
// lane<16: row M=lane, K = {0..7} then {16..23}; lane>=16: K = {8..15} then {24..31}
__device__ __forceinline__ bf16x16 loadA(const unsigned short* __restrict__ row,
                                         int lane, int kstep) {
    const unsigned short* p = row + kstep * 32 + ((lane >> 4) & 1) * 8;
    V16BF t;
    t.q[0] = *(const uint4*)(p);
    t.q[1] = *(const uint4*)(p + 16);
    return t.v;
}

// B-matrix (32x16 bf16) fragment: lane holds column N=lane&15;
// lanes 0-15 hold K={0..7,16..23}, lanes 16-31 hold K={8..15,24..31} (same as A).
// wrow points at the bf16 weight row (D = X @ W^T => B[k][n] = W[n][k]).
__device__ __forceinline__ bf16x16 loadB(const unsigned short* __restrict__ wrow,
                                         int lane, int kstep) {
    const unsigned short* p = wrow + kstep * 32 + ((lane >> 4) & 1) * 8;
    V16BF t;
    t.q[0] = *(const uint4*)(p);
    t.q[1] = *(const uint4*)(p + 16);
    return t.v;
}

// ---------------- K0: generic fp32 -> bf16 convert (weights, one-time) ----
__global__ __launch_bounds__(256) void k_f32_to_bf16(
    const float* __restrict__ src, unsigned short* __restrict__ dst, int n)
{
    int t = blockIdx.x * blockDim.x + threadIdx.x;
    if (t < n) dst[t] = f2bf_bits(src[t]);
}

// ---------------- K1: NCHW fp32 -> (n,L,C) bf16 ----------------
__global__ __launch_bounds__(256) void k_transpose_bf(
    const float* __restrict__ x, unsigned short* __restrict__ inp_bf,
    int L, int total)
{
    int t = blockIdx.x * blockDim.x + threadIdx.x;
    if (t >= total) return;
    int l    = t % L;
    int rest = t / L;
    int c    = rest & (CCH - 1);
    int n    = rest / CCH;
    inp_bf[((size_t)n * L + l) * CCH + c] = f2bf_bits(x[t]);
}

// ---------------- K2: fused value + offset/mask GEMMs ----------------
// v_out (NL,64) fp32, om_out (NL,112) fp32. One wave per 16-pixel row tile.
__global__ __launch_bounds__(256) void k_gemm_vom(
    const unsigned short* __restrict__ inp_bf,
    const unsigned short* __restrict__ value_wbf, const float* __restrict__ value_b,
    const unsigned short* __restrict__ om_wbf,    const float* __restrict__ om_b,
    float* __restrict__ v_out, float* __restrict__ om_out, int NL)
{
    int wave = (int)((blockIdx.x * blockDim.x + threadIdx.x) >> 5);
    int lane = threadIdx.x & 31;
    int row0 = wave * 16;
    if (row0 >= NL) return;  // wave-uniform

    const unsigned short* arow = inp_bf + (size_t)(row0 + (lane & 15)) * CCH;
    bf16x16 a0 = loadA(arow, lane, 0);   // K 0..31
    bf16x16 a1 = loadA(arow, lane, 1);   // K 32..63

    int mbase = (lane >> 4) * 8;
    int ncol  = lane & 15;

    // value projection: 4 N-tiles of 16
#pragma unroll
    for (int t = 0; t < 4; ++t) {
        int n = t * 16 + ncol;
        const unsigned short* wrow = value_wbf + n * CCH;
        bf16x16 b0 = loadB(wrow, lane, 0);
        bf16x16 b1 = loadB(wrow, lane, 1);
        float bias = value_b[n];
        floatx8 c;
#pragma unroll
        for (int i = 0; i < 8; ++i) c[i] = bias;
        c = __builtin_amdgcn_wmma_f32_16x16x32_bf16(false, a0, false, b0, (short)0, c, false, false);
        c = __builtin_amdgcn_wmma_f32_16x16x32_bf16(false, a1, false, b1, (short)0, c, false, false);
#pragma unroll
        for (int r = 0; r < 8; ++r)
            v_out[(size_t)(row0 + mbase + r) * CCH + n] = c[r];
    }

    // offset+mask projection: 7 N-tiles of 16 (112 outputs)
#pragma unroll
    for (int t = 0; t < 7; ++t) {
        int n = t * 16 + ncol;
        const unsigned short* wrow = om_wbf + n * CCH;
        bf16x16 b0 = loadB(wrow, lane, 0);
        bf16x16 b1 = loadB(wrow, lane, 1);
        float bias = om_b[n];
        floatx8 c;
#pragma unroll
        for (int i = 0; i < 8; ++i) c[i] = bias;
        c = __builtin_amdgcn_wmma_f32_16x16x32_bf16(false, a0, false, b0, (short)0, c, false, false);
        c = __builtin_amdgcn_wmma_f32_16x16x32_bf16(false, a1, false, b1, (short)0, c, false, false);
#pragma unroll
        for (int r = 0; r < 8; ++r)
            om_out[(size_t)(row0 + mbase + r) * OMD + n] = c[r];
    }
}

// ---------------- K3: deformable bilinear sampling core ----------------
__device__ __forceinline__ void corner_acc(const float* __restrict__ vpix,
                                           float w, float* __restrict__ acc) {
    const float4* p = (const float4*)vpix;   // 64B-aligned: group base is 16-float aligned
#pragma unroll
    for (int q = 0; q < 4; ++q) {
        float4 f = p[q];
        acc[4 * q + 0] += w * f.x;
        acc[4 * q + 1] += w * f.y;
        acc[4 * q + 2] += w * f.z;
        acc[4 * q + 3] += w * f.w;
    }
}

__global__ __launch_bounds__(256) void k_dcn_core(
    const float* __restrict__ v, const float* __restrict__ om,
    unsigned short* __restrict__ samp_bf, int Nn, int Hh, int Ww)
{
    int L = Hh * Ww;
    int t = blockIdx.x * blockDim.x + threadIdx.x;
    if (t >= Nn * L * GRP) return;
    int g = t & (GRP - 1);
    int r = t >> 2;
    int l = r % L;
    int n = r / L;

    float py = (float)(l / Ww);
    float px = (float)(l % Ww);

    const float* omp  = om + ((size_t)n * L + l) * OMD + g * (3 * KPTS);
    const float* vimg = v + (size_t)n * L * CCH + g * GCH;

    float acc[GCH];
#pragma unroll
    for (int i = 0; i < GCH; ++i) acc[i] = 0.f;

#pragma unroll
    for (int k = 0; k < KPTS; ++k) {
        float offx = omp[2 * k];
        float offy = omp[2 * k + 1];
        float msk  = omp[2 * KPTS + k];
        float sy = py + (float)(k / 3 - 1) + offy;
        float sx = px + (float)(k % 3 - 1) + offx;
        float y0f = floorf(sy), x0f = floorf(sx);
        float wy = sy - y0f, wx = sx - x0f;
        int y0 = (int)y0f, x0 = (int)x0f;
        float w00 = msk * (1.f - wy) * (1.f - wx);
        float w01 = msk * (1.f - wy) * wx;
        float w10 = msk * wy * (1.f - wx);
        float w11 = msk * wy * wx;
        if ((unsigned)y0 < (unsigned)Hh) {
            const float* rowp = vimg + (size_t)(y0 * Ww) * CCH;
            if ((unsigned)x0       < (unsigned)Ww) corner_acc(rowp + (size_t)x0       * CCH, w00, acc);
            if ((unsigned)(x0 + 1) < (unsigned)Ww) corner_acc(rowp + (size_t)(x0 + 1) * CCH, w01, acc);
        }
        if ((unsigned)(y0 + 1) < (unsigned)Hh) {
            const float* rowp = vimg + (size_t)((y0 + 1) * Ww) * CCH;
            if ((unsigned)x0       < (unsigned)Ww) corner_acc(rowp + (size_t)x0       * CCH, w10, acc);
            if ((unsigned)(x0 + 1) < (unsigned)Ww) corner_acc(rowp + (size_t)(x0 + 1) * CCH, w11, acc);
        }
    }

    // pack to bf16 (input of final WMMA GEMM)
    unsigned* o = (unsigned*)(samp_bf + ((size_t)n * L + l) * CCH + g * GCH);
#pragma unroll
    for (int q = 0; q < 8; ++q)
        o[q] = (unsigned)f2bf_bits(acc[2 * q]) |
               ((unsigned)f2bf_bits(acc[2 * q + 1]) << 16);
}

// ---------------- K4: output projection GEMM, write NCHW ----------------
__global__ __launch_bounds__(256) void k_gemm_out(
    const unsigned short* __restrict__ samp_bf,
    const unsigned short* __restrict__ out_wbf, const float* __restrict__ out_b,
    float* __restrict__ y, int NL, int L)
{
    int wave = (int)((blockIdx.x * blockDim.x + threadIdx.x) >> 5);
    int lane = threadIdx.x & 31;
    int row0 = wave * 16;
    if (row0 >= NL) return;  // wave-uniform

    const unsigned short* arow = samp_bf + (size_t)(row0 + (lane & 15)) * CCH;
    bf16x16 a0 = loadA(arow, lane, 0);
    bf16x16 a1 = loadA(arow, lane, 1);

    int n_img = row0 / L;
    int lbase = row0 % L;          // multiple of 16, tiles never cross image boundary
    int mbase = (lane >> 4) * 8;
    int ncol  = lane & 15;

#pragma unroll
    for (int t = 0; t < 4; ++t) {
        int ch = t * 16 + ncol;
        const unsigned short* wrow = out_wbf + ch * CCH;
        bf16x16 b0 = loadB(wrow, lane, 0);
        bf16x16 b1 = loadB(wrow, lane, 1);
        float bias = out_b[ch];
        floatx8 c;
#pragma unroll
        for (int i = 0; i < 8; ++i) c[i] = bias;
        c = __builtin_amdgcn_wmma_f32_16x16x32_bf16(false, a0, false, b0, (short)0, c, false, false);
        c = __builtin_amdgcn_wmma_f32_16x16x32_bf16(false, a1, false, b1, (short)0, c, false, false);

        // D column ch, rows = consecutive pixels -> 8 consecutive floats in NCHW
        float* dst = y + ((size_t)n_img * CCH + ch) * L + lbase + mbase;
        float4 s0 = make_float4(c[0], c[1], c[2], c[3]);
        float4 s1 = make_float4(c[4], c[5], c[6], c[7]);
        *(float4*)(dst)     = s0;
        *(float4*)(dst + 4) = s1;
    }
}

// ---------------- launcher ----------------
extern "C" void kernel_launch(void* const* d_in, const int* in_sizes, int n_in,
                              void* d_out, int out_size, void* d_ws, size_t ws_size,
                              hipStream_t stream) {
    const float* x       = (const float*)d_in[0];
    const float* value_w = (const float*)d_in[1];
    const float* value_b = (const float*)d_in[2];
    const float* om_w    = (const float*)d_in[3];
    const float* om_b    = (const float*)d_in[4];
    const float* out_w   = (const float*)d_in[5];
    const float* out_b   = (const float*)d_in[6];
    float* y = (float*)d_out;

    const int Nn = 4, Hh = 128, Ww = 128;
    const int L  = Hh * Ww;        // 16384
    const int NL = Nn * L;         // 65536

    // workspace carve-up (all offsets stay 256B aligned)
    char* ws = (char*)d_ws;
    size_t off = 0;
    unsigned short* inp_bf  = (unsigned short*)(ws + off); off += (size_t)NL * CCH * sizeof(unsigned short); // 8 MB
    float*          v_buf   = (float*)(ws + off);          off += (size_t)NL * CCH * sizeof(float);          // 16 MB
    float*          om_buf  = (float*)(ws + off);          off += (size_t)NL * OMD * sizeof(float);          // 28 MB
    unsigned short* samp_bf = (unsigned short*)(ws + off); off += (size_t)NL * CCH * sizeof(unsigned short); // 8 MB
    unsigned short* vw_bf   = (unsigned short*)(ws + off); off += (size_t)CCH * CCH * sizeof(unsigned short);
    unsigned short* omw_bf  = (unsigned short*)(ws + off); off += (size_t)OMD * CCH * sizeof(unsigned short);
    unsigned short* outw_bf = (unsigned short*)(ws + off); off += (size_t)CCH * CCH * sizeof(unsigned short);
    (void)ws_size;

    // K0: one-time weight conversion to bf16 (30 KB total, stays in L2)
    k_f32_to_bf16<<<(CCH * CCH + 255) / 256, 256, 0, stream>>>(value_w, vw_bf,  CCH * CCH);
    k_f32_to_bf16<<<(OMD * CCH + 255) / 256, 256, 0, stream>>>(om_w,    omw_bf, OMD * CCH);
    k_f32_to_bf16<<<(CCH * CCH + 255) / 256, 256, 0, stream>>>(out_w,   outw_bf, CCH * CCH);

    // K1: transpose + convert input to bf16
    {
        int total  = NL * CCH;                       // 4,194,304
        int blocks = (total + 255) / 256;
        k_transpose_bf<<<blocks, 256, 0, stream>>>(x, inp_bf, L, total);
    }
    // K2: value + offset/mask projections (WMMA)
    {
        int waves  = NL / 16;                        // 4096 row tiles
        int blocks = waves / 8;                      // 8 waves per 256-thread block
        k_gemm_vom<<<blocks, 256, 0, stream>>>(inp_bf, vw_bf, value_b,
                                               omw_bf, om_b, v_buf, om_buf, NL);
    }
    // K3: deformable bilinear sampling
    {
        int total  = NL * GRP;                       // 262,144
        int blocks = (total + 255) / 256;
        k_dcn_core<<<blocks, 256, 0, stream>>>(v_buf, om_buf, samp_bf, Nn, Hh, Ww);
    }
    // K4: output projection (WMMA) -> NCHW fp32
    {
        int waves  = NL / 16;
        int blocks = waves / 8;
        k_gemm_out<<<blocks, 256, 0, stream>>>(samp_bf, outw_bf, out_b, y, NL, L);
    }
}
